// GInNER_33054068310314
// MI455X (gfx1250) — matrix operand order repaired and checked by reading, not compiled, not run on hardware
//
#include <hip/hip_runtime.h>

typedef __attribute__((ext_vector_type(16))) _Float16 v16h;
typedef __attribute__((ext_vector_type(8)))  float    v8f;

#define NN   4096
#define FIN  768
#define FH   64
#define NH   8
#define NC   9
#define NCP  16
#define ALPHA 0.2f

// ---------------------------------------------------------------------------
// WMMA fragment packing helpers (v_wmma_f32_16x16x32_f16 operand layouts).
// A (16x32 f16): lane = r + 16*half, element e: K = (e/8)*16 + half*8 + (e%8)
// B (32x16 f16): lane = c + 16*half, element e: K = (e/8)*16 + half*8 + (e%8)
// Fragments stored [tile][lane][e]: each lane's 16 f16 contiguous -> 2x b128.
// ---------------------------------------------------------------------------
__device__ __forceinline__ size_t packA_idx(int i, int j, int KT) {
    int tm = i >> 4, r = i & 15;
    int kt = j >> 5, kj = j & 31;
    int half = (kj >> 3) & 1;
    int e = ((kj >> 4) << 3) | (kj & 7);
    int lane = r + (half << 4);
    return ((size_t)(tm * KT + kt) * 32 + lane) * 16 + e;
}

// mask[i][j] = (adj[i][j]>0 || adj[j][i]>0 || i==j)
__global__ void build_mask(const float* __restrict__ adj, unsigned char* __restrict__ mask) {
    int idx = blockIdx.x * 256 + threadIdx.x;
    int i = idx >> 12, j = idx & (NN - 1);
    float a = adj[(size_t)i * NN + j];
    float b = adj[(size_t)j * NN + i];
    mask[idx] = (a > 0.f || b > 0.f || i == j) ? (unsigned char)1 : (unsigned char)0;
}

// row-sum normalize features, emit packed A fragments (KT = FIN/32)
__global__ __launch_bounds__(256) void norm_feats(const float* __restrict__ xin,
                                                  _Float16* __restrict__ xAp) {
    __shared__ float red[256];
    int row = blockIdx.x, t = threadIdx.x;
    float s = 0.f;
    for (int c = t; c < FIN; c += 256) s += xin[(size_t)row * FIN + c];
    red[t] = s; __syncthreads();
    for (int o = 128; o > 0; o >>= 1) { if (t < o) red[t] += red[t + o]; __syncthreads(); }
    float rs = red[0];
    float inv = (rs != 0.f) ? 1.f / rs : 0.f;
    for (int c = t; c < FIN; c += 256)
        xAp[packA_idx(row, c, FIN / 32)] = (_Float16)(xin[(size_t)row * FIN + c] * inv);
}

// Pack f32 [K][stride] (cols < Nv valid, rest zero) into B fragments [K][N], N%16==0
__global__ void pack_b_f32(const float* __restrict__ src, _Float16* __restrict__ Bp,
                           int K, int N, int Nv, int stride) {
    int idx = blockIdx.x * 256 + threadIdx.x;
    if (idx >= K * N) return;
    int e    = idx & 15;
    int lane = (idx >> 4) & 31;
    int frag = idx >> 9;                 // tn*KT + kt
    int KT = K >> 5;
    int tn = frag / KT, kt = frag % KT;
    int half = lane >> 4, c = lane & 15;
    int k = kt * 32 + ((e >> 3) << 4) + (half << 3) + (e & 7);
    int n = tn * 16 + c;
    float v = (n < Nv) ? src[(size_t)k * stride + n] : 0.f;
    Bp[idx] = (_Float16)v;
}

// ---------------------------------------------------------------------------
// Packed WMMA GEMM: C[M, NT*16] = A[M, KT*32] * B. 4 waves/block, each wave
// owns a 16x(NT*16) strip. Unroll-by-2 ping-pong buffers (a0/b0, a1/b1): no
// loop-carried register copies, loads stay one WMMA-group ahead. KT must be
// even (all call sites: 24, 128, 16).
// ---------------------------------------------------------------------------
template <int NT>
__global__ __launch_bounds__(128) void wmma_gemm_packed(const v16h* __restrict__ Ap,
                                                        const v16h* __restrict__ Bp,
                                                        float* __restrict__ C, int KT) {
    int wave = threadIdx.x >> 5;
    int lane = threadIdx.x & 31;
    int tm = blockIdx.x * 4 + wave;

    const v16h* ap = Ap + (size_t)tm * KT * 32 + lane;
    const v16h* bp = Bp + lane;

    v8f zero = {};
    v8f acc[NT];
    #pragma unroll
    for (int nt = 0; nt < NT; ++nt) acc[nt] = zero;

    v16h a0, a1;
    v16h b0[NT], b1[NT];

    // prologue: fragments for kt = 0
    a0 = ap[0];
    #pragma unroll
    for (int nt = 0; nt < NT; ++nt) b0[nt] = bp[(size_t)nt * KT * 32];

    int kt = 0;
    for (; kt < KT - 2; kt += 2) {
        a1 = ap[(size_t)(kt + 1) * 32];
        #pragma unroll
        for (int nt = 0; nt < NT; ++nt) b1[nt] = bp[(size_t)(nt * KT + kt + 1) * 32];

        #pragma unroll
        for (int nt = 0; nt < NT; ++nt)
            acc[nt] = __builtin_amdgcn_wmma_f32_16x16x32_f16(false, a0, false, b0[nt],
                                                             (short)0, acc[nt], false, false);

        a0 = ap[(size_t)(kt + 2) * 32];
        #pragma unroll
        for (int nt = 0; nt < NT; ++nt) b0[nt] = bp[(size_t)(nt * KT + kt + 2) * 32];

        #pragma unroll
        for (int nt = 0; nt < NT; ++nt)
            acc[nt] = __builtin_amdgcn_wmma_f32_16x16x32_f16(false, a1, false, b1[nt],
                                                             (short)0, acc[nt], false, false);
    }

    // epilogue: kt == KT-2
    a1 = ap[(size_t)(kt + 1) * 32];
    #pragma unroll
    for (int nt = 0; nt < NT; ++nt) b1[nt] = bp[(size_t)(nt * KT + kt + 1) * 32];
    #pragma unroll
    for (int nt = 0; nt < NT; ++nt)
        acc[nt] = __builtin_amdgcn_wmma_f32_16x16x32_f16(false, a0, false, b0[nt],
                                                         (short)0, acc[nt], false, false);
    #pragma unroll
    for (int nt = 0; nt < NT; ++nt)
        acc[nt] = __builtin_amdgcn_wmma_f32_16x16x32_f16(false, a1, false, b1[nt],
                                                         (short)0, acc[nt], false, false);

    int half = lane >> 4, l16 = lane & 15;
    const int N = NT * 16;
    #pragma unroll
    for (int nt = 0; nt < NT; ++nt) {
        #pragma unroll
        for (int v = 0; v < 8; ++v)
            C[(size_t)(tm * 16 + v + (half << 3)) * N + nt * 16 + l16] = acc[nt][v];
    }
}

// f1[i] = h[i,:Fv] . a[0:Fv]; f2[i] = h[i,:Fv] . a[Fv:2Fv]
__global__ void att_scores(const float* __restrict__ h, const float* __restrict__ a,
                           float* __restrict__ f1, float* __restrict__ f2,
                           int rows, int F, int Fv) {
    int r = blockIdx.x * 256 + threadIdx.x;
    if (r >= rows) return;
    float s1 = 0.f, s2 = 0.f;
    for (int c = 0; c < Fv; ++c) {
        float v = h[(size_t)r * F + c];
        s1 += v * a[c];
        s2 += v * a[Fv + c];
    }
    f1[r] = s1; f2[r] = s2;
}

// Masked LeakyReLU + row softmax; writes attention row as packed A fragments
__global__ __launch_bounds__(256) void att_softmax(const float* __restrict__ f1,
                                                   const float* __restrict__ f2,
                                                   const unsigned char* __restrict__ mask,
                                                   _Float16* __restrict__ attAp) {
    __shared__ float ebuf[NN];
    __shared__ float red[256];
    int i = blockIdx.x, t = threadIdx.x;
    float fi = f1[i];
    const unsigned char* mrow = mask + (size_t)i * NN;

    float mx = -3.0e38f;
    for (int j = t; j < NN; j += 256) {
        float e;
        if (mrow[j]) {
            float v = fi + f2[j];
            e = (v > 0.f) ? v : ALPHA * v;
        } else {
            e = -3.0e38f;
        }
        ebuf[j] = e;
        mx = fmaxf(mx, e);
    }
    red[t] = mx; __syncthreads();
    for (int o = 128; o > 0; o >>= 1) { if (t < o) red[t] = fmaxf(red[t], red[t + o]); __syncthreads(); }
    mx = red[0]; __syncthreads();

    float sum = 0.f;
    for (int j = t; j < NN; j += 256) {
        float e = ebuf[j];
        float x = (e > -1.0e38f) ? __expf(e - mx) : 0.f;
        ebuf[j] = x;
        sum += x;
    }
    red[t] = sum; __syncthreads();
    for (int o = 128; o > 0; o >>= 1) { if (t < o) red[t] += red[t + o]; __syncthreads(); }
    float inv = 1.f / red[0];

    for (int j = t; j < NN; j += 256)
        attAp[packA_idx(i, j, NN / 32)] = (_Float16)(ebuf[j] * inv);
}

// ELU + transpose-concat: hatt [H][N][FH] -> x1 packed A fragments (K = H*FH = 512)
__global__ void elu_concat(const float* __restrict__ hatt, _Float16* __restrict__ x1Ap) {
    int idx = blockIdx.x * 256 + threadIdx.x;  // H*N*FH
    float v = hatt[idx];
    float e = (v > 0.f) ? v : (__expf(v) - 1.f);
    int f = idx % FH;
    int tmp = idx / FH;
    int i  = tmp % NN;
    int hd = tmp / NN;
    x1Ap[packA_idx(i, hd * FH + f, (NH * FH) / 32)] = (_Float16)e;
}

// elu(elu(.)) then softmax over NC classes
__global__ void final_softmax(const float* __restrict__ o2, float* __restrict__ out) {
    int i = blockIdx.x * 256 + threadIdx.x;
    if (i >= NN) return;
    float v[NC];
    float mx = -3.0e38f;
    #pragma unroll
    for (int c = 0; c < NC; ++c) {
        float x = o2[(size_t)i * NCP + c];
        x = (x > 0.f) ? x : (__expf(x) - 1.f);
        x = (x > 0.f) ? x : (__expf(x) - 1.f);
        v[c] = x;
        mx = fmaxf(mx, x);
    }
    float s = 0.f;
    #pragma unroll
    for (int c = 0; c < NC; ++c) { v[c] = __expf(v[c] - mx); s += v[c]; }
    float inv = 1.f / s;
    #pragma unroll
    for (int c = 0; c < NC; ++c) out[(size_t)i * NC + c] = v[c] * inv;
}

// ---------------------------------------------------------------------------
extern "C" void kernel_launch(void* const* d_in, const int* in_sizes, int n_in,
                              void* d_out, int out_size, void* d_ws, size_t ws_size,
                              hipStream_t stream) {
    const float* input   = (const float*)d_in[0];  // [N, FIN]
    const float* adj     = (const float*)d_in[1];  // [N, N]
    const float* W_heads = (const float*)d_in[2];  // [H, FIN, FH]
    const float* a_heads = (const float*)d_in[3];  // [H, 2*FH]
    const float* W_out   = (const float*)d_in[4];  // [H*FH, C]
    const float* a_out   = (const float*)d_in[5];  // [2*C]
    float* out = (float*)d_out;                    // [N, C]

    char* p = (char*)d_ws;
    auto carve = [&](size_t bytes) -> char* {
        char* r = p;
        p += (bytes + 255) & ~(size_t)255;
        return r;
    };
    unsigned char* mask  = (unsigned char*)carve((size_t)NN * NN);
    _Float16* xAp   = (_Float16*)carve((size_t)NN * FIN * 2);          // A frags, KT=24
    _Float16* WbP   = (_Float16*)carve((size_t)NH * FIN * FH * 2);     // B frags per head
    float*    h     = (float*)   carve((size_t)NH * NN * FH * 4);
    _Float16* hhP   = (_Float16*)carve((size_t)NH * NN * FH * 2);      // B frags per head
    float*    f1    = (float*)   carve((size_t)NH * NN * 4);
    float*    f2    = (float*)   carve((size_t)NH * NN * 4);
    _Float16* attAp = (_Float16*)carve((size_t)NN * NN * 2);           // A frags, KT=128
    float*    hatt  = (float*)   carve((size_t)NH * NN * FH * 4);
    _Float16* x1Ap  = (_Float16*)carve((size_t)NN * NH * FH * 2);      // A frags, KT=16
    _Float16* WoP   = (_Float16*)carve((size_t)(NH * FH) * NCP * 2);   // B frags
    float*    h2    = (float*)   carve((size_t)NN * NCP * 4);
    _Float16* h2P   = (_Float16*)carve((size_t)NN * NCP * 2);          // B frags
    float*    f1o   = (float*)   carve((size_t)NN * 4);
    float*    f2o   = (float*)   carve((size_t)NN * 4);
    float*    o2    = (float*)   carve((size_t)NN * NCP * 4);
    (void)ws_size; (void)in_sizes; (void)n_in; (void)out_size;

    // 1) mask, normalized packed features, packed head weights
    build_mask<<<(NN * NN) / 256, 256, 0, stream>>>(adj, mask);
    norm_feats<<<NN, 256, 0, stream>>>(input, xAp);
    for (int hd = 0; hd < NH; ++hd)
        pack_b_f32<<<(FIN * FH) / 256, 256, 0, stream>>>(
            W_heads + (size_t)hd * FIN * FH, WbP + (size_t)hd * FIN * FH, FIN, FH, FH, FH);

    // 2) per-head feature GEMM: h = x @ W  (M=4096, K=768, N=64)
    for (int hd = 0; hd < NH; ++hd)
        wmma_gemm_packed<4><<<NN / 64, 128, 0, stream>>>(
            (const v16h*)xAp, (const v16h*)(WbP + (size_t)hd * FIN * FH),
            h + (size_t)hd * NN * FH, FIN / 32);

    // 3) attention scores + pack h as B fragments for att @ h
    for (int hd = 0; hd < NH; ++hd) {
        att_scores<<<(NN + 255) / 256, 256, 0, stream>>>(
            h + (size_t)hd * NN * FH, a_heads + (size_t)hd * 2 * FH,
            f1 + (size_t)hd * NN, f2 + (size_t)hd * NN, NN, FH, FH);
        pack_b_f32<<<(NN * FH) / 256, 256, 0, stream>>>(
            h + (size_t)hd * NN * FH, hhP + (size_t)hd * NN * FH, NN, FH, FH, FH);
    }

    // 4) per head: masked softmax (packed A) then att @ h  (M=4096, K=4096, N=64)
    for (int hd = 0; hd < NH; ++hd) {
        att_softmax<<<NN, 256, 0, stream>>>(f1 + (size_t)hd * NN, f2 + (size_t)hd * NN,
                                            mask, attAp);
        wmma_gemm_packed<4><<<NN / 64, 128, 0, stream>>>(
            (const v16h*)attAp, (const v16h*)(hhP + (size_t)hd * NN * FH),
            hatt + (size_t)hd * NN * FH, NN / 32);
    }

    // 5) ELU + concat -> x1 packed A fragments
    elu_concat<<<(NH * NN * FH) / 256, 256, 0, stream>>>(hatt, x1Ap);

    // 6) out layer
    pack_b_f32<<<((NH * FH) * NCP) / 256, 256, 0, stream>>>(W_out, WoP, NH * FH, NCP, NC, NC);
    wmma_gemm_packed<1><<<NN / 64, 128, 0, stream>>>(
        (const v16h*)x1Ap, (const v16h*)WoP, h2, (NH * FH) / 32);
    att_scores<<<(NN + 255) / 256, 256, 0, stream>>>(h2, a_out, f1o, f2o, NN, NCP, NC);
    pack_b_f32<<<(NN * NCP) / 256, 256, 0, stream>>>(h2, h2P, NN, NCP, NCP, NCP);
    att_softmax<<<NN, 256, 0, stream>>>(f1o, f2o, mask, attAp);
    wmma_gemm_packed<1><<<NN / 64, 128, 0, stream>>>(
        (const v16h*)attAp, (const v16h*)h2P, o2, NN / 32);

    // 7) elu(elu(.)) + class softmax
    final_softmax<<<(NN + 255) / 256, 256, 0, stream>>>(o2, out);
}
